// MEGNet_36301063586431
// MI455X (gfx1250) — matrix operand order repaired and checked by reading, not compiled, not run on hardware
//
#include <hip/hip_runtime.h>

// ---------------------------------------------------------------------------
// MEGNet forward for gfx1250 (CDNA5). All GEMMs run through
// v_wmma_f32_16x16x32_bf16. Activations carry a bf16 shadow copy in memory so
// the GEMM inner loop is pure b128 loads + WMMA (no conversion, no guards).
// NSEG is a compile-time template param so gather pointers hoist out of the
// K-loop (segments are 64 wide = exactly 2 K-blocks each).
// ---------------------------------------------------------------------------

#define NN 40000
#define NE 400000
#define NG 256

typedef __attribute__((ext_vector_type(16))) __bf16 v16bf;
typedef __attribute__((ext_vector_type(8))) unsigned v8u;
typedef __attribute__((ext_vector_type(8))) float v8f;
typedef unsigned short ushort_t;

__device__ __forceinline__ unsigned short f2bf(float f) {
  unsigned u = __builtin_bit_cast(unsigned, f);
  unsigned r = u + 0x7fffu + ((u >> 16) & 1u);   // round-to-nearest-even
  return (unsigned short)(r >> 16);
}

__device__ __forceinline__ unsigned pk2bf(float a, float b) {
  return (unsigned)f2bf(a) | ((unsigned)f2bf(b) << 16);
}

struct SegA {
  const ushort_t* p[4];  // bf16 segment base pointers (64 cols wide if NSEG>1,
                         // strideA cols wide if NSEG==1)
  const int* ix[4];      // per-row gather index (nullptr => identity)
};

// C[M,64] = act(A[M,K] @ W[K,64] + bias).
// A is bf16, zero-padded to strideA (multiple of 32) when NSEG==1.
// kblocks = K-loop trip count; W rows >= K are staged as zero.
template <int NSEG>
__global__ __launch_bounds__(256) void gemm_bf16_wmma(
    SegA A, int M, int K, int strideA, int kblocks,
    const float* __restrict__ W, const float* __restrict__ bias,
    float* __restrict__ out, ushort_t* __restrict__ outbf, int doRelu)
{
  extern __shared__ unsigned short smem[];   // [kb][nt][lane][16] bf16 frags
  const int tid = threadIdx.x;

  // Stage W into LDS in exact per-lane B-fragment layout (bf16).
  const int nfrag = kblocks * 128;           // kblocks * 4 ntiles * 32 lanes
  for (int f = tid; f < nfrag; f += 256) {
    const int kb = f >> 7;
    const int nt = (f >> 5) & 3;
    const int ln = f & 31;
    const int n = nt * 16 + (ln & 15);
    const int kbase = kb * 32 + ((ln >> 4) << 4);   // lane half picks K range
    unsigned* dst = (unsigned*)&smem[f * 16];
#pragma unroll
    for (int j = 0; j < 8; ++j) {
      const int k0 = kbase + 2 * j;
      const float w0 = (k0 < K) ? W[k0 * 64 + n] : 0.0f;
      const float w1 = (k0 + 1 < K) ? W[(k0 + 1) * 64 + n] : 0.0f;
      dst[j] = pk2bf(w0, w1);
    }
  }
  __syncthreads();

  const int wave = tid >> 5;
  const int lane = tid & 31;
  const int tileM = blockIdx.x * 8 + wave;
  if (tileM * 16 >= M) return;
  const int m = tileM * 16 + (lane & 15);
  const int half = lane >> 4;

  v8f acc[4] = {};

  // One MMA step: 32 K-values. chunkbase points at this lane's row, k-offset 0.
  auto mma_step = [&](const ushort_t* chunkbase, int kb) {
    v8u au;
#pragma unroll
    for (int ch = 0; ch < 2; ++ch) {
      const uint4 q = *(const uint4*)(chunkbase + half * 16 + ch * 8);
      au[ch * 4 + 0] = q.x; au[ch * 4 + 1] = q.y;
      au[ch * 4 + 2] = q.z; au[ch * 4 + 3] = q.w;
    }
    const v16bf af = __builtin_bit_cast(v16bf, au);
    const unsigned short* bb = &smem[(kb * 128 + lane) * 16];
#pragma unroll
    for (int nt = 0; nt < 4; ++nt) {
      const v16bf bf = *(const v16bf*)(bb + nt * 512);
      acc[nt] = __builtin_amdgcn_wmma_f32_16x16x32_bf16(
          false, af, false, bf, (short)0, acc[nt], false, false);
    }
  };

  if constexpr (NSEG == 1) {
    const ushort_t* rowp = A.p[0] + (long long)m * strideA;
    for (int kb = 0; kb < kblocks; ++kb)
      mma_step(rowp + kb * 32, kb);
  } else {
    // All gather indices / row pointers resolved up front (hoisted).
    const ushort_t* rowp[NSEG];
#pragma unroll
    for (int s = 0; s < NSEG; ++s) {
      const int* ixp = A.ix[s];
      const int ridx = ixp ? ixp[m] : m;
      rowp[s] = A.p[s] + (long long)ridx * 64;
    }
#pragma unroll
    for (int s = 0; s < NSEG; ++s) {
#pragma unroll
      for (int h = 0; h < 2; ++h)
        mma_step(rowp[s] + h * 32, s * 2 + h);
    }
  }

  // Epilogue: C layout — VGPR v holds row m = half*8 + v, col n = nt*16+lane%16
#pragma unroll
  for (int nt = 0; nt < 4; ++nt) {
    const int n = nt * 16 + (lane & 15);
    const float bv = bias ? bias[n] : 0.0f;
#pragma unroll
    for (int v = 0; v < 8; ++v) {
      const long long mr = tileM * 16 + half * 8 + v;
      float val = acc[nt][v] + bv;
      if (doRelu) val = fmaxf(val, 0.0f);
      out[mr * 64 + n] = val;
      if (outbf) outbf[mr * 64 + n] = f2bf(val);
    }
  }
}

// ------------------------------- utility kernels ---------------------------

__global__ void k_zero(float* p, long long n) {
  long long i = (long long)blockIdx.x * 256 + threadIdx.x;
  if (i < n) p[i] = 0.0f;
}

__global__ void k_add(float* dst, const float* a, const float* b,
                      ushort_t* shadow, long long n) {
  long long i = (long long)blockIdx.x * 256 + threadIdx.x;
  if (i < n) {
    const float v = a[i] + b[i];
    dst[i] = v;
    shadow[i] = f2bf(v);
  }
}

__global__ void k_gather_int(const int* batch, const int* row, int* ubr, int E) {
  int e = blockIdx.x * 256 + threadIdx.x;
  if (e < E) ubr[e] = batch[row[e]];
}

// f32 [rows,K] -> zero-padded bf16 [rows,Kpad]
__global__ void k_cvt_pad(const float* __restrict__ src, int K, int Kpad,
                          ushort_t* __restrict__ dst, long long rows) {
  long long t = (long long)blockIdx.x * 256 + threadIdx.x;
  if (t >= rows * Kpad) return;
  const long long r = t / Kpad;
  const int c = (int)(t - r * Kpad);
  dst[t] = (c < K) ? f2bf(src[r * K + c]) : (ushort_t)0;
}

// BatchNorm stats: accum[0:64]=sum, accum[64:128]=sumsq (per column)
__global__ void k_bn_stats(const float* __restrict__ Y, long long M,
                           float* __restrict__ accum) {
  __shared__ float ls[256], ls2[256];
  const int tid = threadIdx.x;
  const int col = tid & 63;
  const int rg = tid >> 6;
  float s = 0.0f, s2 = 0.0f;
  for (long long r = (long long)blockIdx.x * 4 + rg; r < M;
       r += (long long)gridDim.x * 4) {
    const float v = Y[r * 64 + col];
    s += v; s2 += v * v;
  }
  ls[tid] = s; ls2[tid] = s2;
  __syncthreads();
  if (tid < 64) {
    float a = ls[tid] + ls[tid + 64] + ls[tid + 128] + ls[tid + 192];
    float b = ls2[tid] + ls2[tid + 64] + ls2[tid + 128] + ls2[tid + 192];
    atomicAdd(&accum[tid], a);
    atomicAdd(&accum[64 + tid], b);
  }
}

__global__ void k_bn_apply(float* __restrict__ Y, long long M,
                           const float* __restrict__ accum,
                           const float* __restrict__ g,
                           const float* __restrict__ beta,
                           ushort_t* __restrict__ shadow) {
  long long i = (long long)blockIdx.x * 256 + threadIdx.x;
  if (i >= M * 64) return;
  const int c = (int)(i & 63);
  const float invM = 1.0f / (float)M;
  const float mu = accum[c] * invM;
  const float var = accum[64 + c] * invM - mu * mu;
  const float sc = g[c] * rsqrtf(var + 1e-5f);
  const float v = (Y[i] - mu) * sc + beta[c];
  Y[i] = v;
  shadow[i] = f2bf(v);
}

__global__ void k_scatter_add(const float* __restrict__ src,
                              const int* __restrict__ idx,
                              float* __restrict__ acc,
                              float* __restrict__ cnt, long long rows) {
  long long t = (long long)blockIdx.x * 256 + threadIdx.x;
  if (t >= rows * 64) return;
  const long long r = t >> 6;
  const int c = (int)(t & 63);
  const int d = idx[r];
  atomicAdd(&acc[(long long)d * 64 + c], src[t]);
  if (c == 0) atomicAdd(&cnt[d], 1.0f);
}

__global__ void k_divide(float* __restrict__ acc, const float* __restrict__ cnt,
                         ushort_t* __restrict__ shadow, long long rows) {
  long long t = (long long)blockIdx.x * 256 + threadIdx.x;
  if (t >= rows * 64) return;
  const float v = acc[t] / fmaxf(cnt[t >> 6], 1.0f);
  acc[t] = v;
  shadow[t] = f2bf(v);
}

// out[g] = h[g,:64] . W[:,0] + b[0]
__global__ void k_final(const float* __restrict__ h, const float* __restrict__ W,
                        const float* __restrict__ b, float* __restrict__ out) {
  int g = blockIdx.x * 256 + threadIdx.x;
  if (g >= NG) return;
  float s = b[0];
#pragma unroll
  for (int k = 0; k < 64; ++k) s += h[g * 64 + k] * W[k];
  out[g] = s;
}

// ------------------------------- host helpers ------------------------------

static inline int gblocks(long long n) { return (int)((n + 255) / 256); }

static void launch_gemm(hipStream_t st,
                        const ushort_t* p0, const int* i0,
                        const ushort_t* p1, const int* i1,
                        const ushort_t* p2, const int* i2,
                        const ushort_t* p3, const int* i3,
                        int nseg, int M, int K, int strideA,
                        const float* W, const float* b,
                        float* out, ushort_t* outbf, int relu) {
  SegA A;
  A.p[0] = p0; A.p[1] = p1; A.p[2] = p2; A.p[3] = p3;
  A.ix[0] = i0; A.ix[1] = i1; A.ix[2] = i2; A.ix[3] = i3;
  const int kblocks = (nseg > 1) ? (K / 32) : (strideA / 32);
  const int grid = (M / 16 + 7) / 8;
  const size_t sh = (size_t)kblocks * 4096;
  switch (nseg) {
    case 1:
      gemm_bf16_wmma<1><<<grid, 256, sh, st>>>(A, M, K, strideA, kblocks, W, b,
                                               out, outbf, relu);
      break;
    case 3:
      gemm_bf16_wmma<3><<<grid, 256, sh, st>>>(A, M, K, strideA, kblocks, W, b,
                                               out, outbf, relu);
      break;
    default:
      gemm_bf16_wmma<4><<<grid, 256, sh, st>>>(A, M, K, strideA, kblocks, W, b,
                                               out, outbf, relu);
      break;
  }
}

static void G1(hipStream_t st, const ushort_t* a, int M, int K, int strideA,
               const float* W, const float* b, float* out, ushort_t* outbf,
               int relu) {
  launch_gemm(st, a, nullptr, nullptr, nullptr, nullptr, nullptr, nullptr,
              nullptr, 1, M, K, strideA, W, b, out, outbf, relu);
}

static void run_bn(hipStream_t st, float* Y, long long M, const float* g,
                   const float* be, float* accum, ushort_t* shadow) {
  k_zero<<<1, 128, 0, st>>>(accum, 128);
  long long sb = (M + 3) / 4; if (sb > 4096) sb = 4096;
  k_bn_stats<<<(int)sb, 256, 0, st>>>(Y, M, accum);
  k_bn_apply<<<gblocks(M * 64), 256, 0, st>>>(Y, M, accum, g, be, shadow);
}

// mean lands in `acc` (in place after divide); bf16 shadow written too
static void run_smean(hipStream_t st, const float* src, const int* idx,
                      long long rows, float* acc, float* cnt, long long outRows,
                      ushort_t* shadow) {
  k_zero<<<gblocks(outRows * 64), 256, 0, st>>>(acc, outRows * 64);
  k_zero<<<gblocks(outRows), 256, 0, st>>>(cnt, outRows);
  k_scatter_add<<<gblocks(rows * 64), 256, 0, st>>>(src, idx, acc, cnt, rows);
  k_divide<<<gblocks(outRows * 64), 256, 0, st>>>(acc, cnt, shadow, outRows);
}

// ------------------------------- driver ------------------------------------

extern "C" void kernel_launch(void* const* d_in, const int* in_sizes, int n_in,
                              void* d_out, int out_size, void* d_ws, size_t ws_size,
                              hipStream_t stream) {
  (void)in_sizes; (void)out_size; (void)ws_size;
  if (n_in < 155) return;   // expected: x, edge_attr, u, 150 param leaves, edge_index, batch

  const float* x         = (const float*)d_in[0];
  const float* edge_attr = (const float*)d_in[1];
  const float* u         = (const float*)d_in[2];

  int pi = 3;
  auto nxt = [&]() { return (const float*)d_in[pi++]; };

  struct Lin { const float* W; const float* b; };
  struct Mdl { const float* g[3]; const float* be[3]; Lin lin[3]; };
  struct GCB { Lin e0, e1, u0, u1, x0, x1; Mdl edge, glob, node; } gc[3];

  // params pytree leaf order (dict keys sorted): gc, out, post, pre.
  // within gc[i]: e_embed, edge(bn,lin), glob(bn,lin), node(bn,lin), u_embed, x_embed
  for (int i = 0; i < 3; ++i) {
    gc[i].e0 = {nxt(), nxt()}; gc[i].e1 = {nxt(), nxt()};
    auto rdm = [&](Mdl& m) {
      for (int s = 0; s < 3; ++s) { m.g[s] = nxt(); m.be[s] = nxt(); }
      for (int s = 0; s < 3; ++s) { m.lin[s] = {nxt(), nxt()}; }
    };
    rdm(gc[i].edge);
    rdm(gc[i].glob);
    rdm(gc[i].node);
    gc[i].u0 = {nxt(), nxt()}; gc[i].u1 = {nxt(), nxt()};
    gc[i].x0 = {nxt(), nxt()}; gc[i].x1 = {nxt(), nxt()};
  }
  Lin outl = {nxt(), nxt()};
  Lin post = {nxt(), nxt()};
  Lin pre  = {nxt(), nxt()};

  const int* edge_index = (const int*)d_in[pi++];
  const int* batch      = (const int*)d_in[pi++];
  const int* row = edge_index;
  const int* col = edge_index + NE;

  // ----- workspace layout (bump allocator, 256B aligned) -----
  char* wp = (char*)d_ws;
  auto alloc = [&](size_t bytes) -> void* {
    void* r = (void*)wp;
    wp += (bytes + 255) & ~(size_t)255;
    return r;
  };
  // f32 activations
  float* xc    = (float*)alloc((size_t)NN * 64 * 4);
  float* x_t   = (float*)alloc((size_t)NN * 64 * 4);
  float* vacc  = (float*)alloc((size_t)NN * 64 * 4);
  float* nA    = (float*)alloc((size_t)NN * 64 * 4);
  float* nB    = (float*)alloc((size_t)NN * 64 * 4);
  float* ec    = (float*)alloc((size_t)NE * 64 * 4);
  float* e_t   = (float*)alloc((size_t)NE * 64 * 4);
  float* eA    = (float*)alloc((size_t)NE * 64 * 4);
  float* eB    = (float*)alloc((size_t)NE * 64 * 4);
  float* ucb   = (float*)alloc((size_t)NG * 64 * 4);
  float* u_t   = (float*)alloc((size_t)NG * 64 * 4);
  float* gue   = (float*)alloc((size_t)NG * 64 * 4);
  float* guv   = (float*)alloc((size_t)NG * 64 * 4);
  float* gA    = (float*)alloc((size_t)NG * 64 * 4);
  float* gB    = (float*)alloc((size_t)NG * 64 * 4);
  float* xpool = (float*)alloc((size_t)NG * 64 * 4);
  float* epool = (float*)alloc((size_t)NG * 64 * 4);
  float* pA    = (float*)alloc((size_t)NG * 64 * 4);
  float* accum = (float*)alloc(128 * 4);
  float* cnt_n = (float*)alloc((size_t)NN * 4);
  float* cnt_g = (float*)alloc((size_t)NG * 4);
  int*   ubr   = (int*)alloc((size_t)NE * 4);
  // bf16 shadows
  ushort_t* xc_bf    = (ushort_t*)alloc((size_t)NN * 64 * 2);
  ushort_t* x_t_bf   = (ushort_t*)alloc((size_t)NN * 64 * 2);
  ushort_t* vacc_bf  = (ushort_t*)alloc((size_t)NN * 64 * 2);
  ushort_t* nA_bf    = (ushort_t*)alloc((size_t)NN * 64 * 2);
  ushort_t* nB_bf    = (ushort_t*)alloc((size_t)NN * 64 * 2);
  ushort_t* ec_bf    = (ushort_t*)alloc((size_t)NE * 64 * 2);
  ushort_t* e_t_bf   = (ushort_t*)alloc((size_t)NE * 64 * 2);
  ushort_t* eA_bf    = (ushort_t*)alloc((size_t)NE * 64 * 2);
  ushort_t* eB_bf    = (ushort_t*)alloc((size_t)NE * 64 * 2);
  ushort_t* ucb_bf   = (ushort_t*)alloc((size_t)NG * 64 * 2);
  ushort_t* u_t_bf   = (ushort_t*)alloc((size_t)NG * 64 * 2);
  ushort_t* gue_bf   = (ushort_t*)alloc((size_t)NG * 64 * 2);
  ushort_t* guv_bf   = (ushort_t*)alloc((size_t)NG * 64 * 2);
  ushort_t* gA_bf    = (ushort_t*)alloc((size_t)NG * 64 * 2);
  ushort_t* gB_bf    = (ushort_t*)alloc((size_t)NG * 64 * 2);
  ushort_t* xpool_bf = (ushort_t*)alloc((size_t)NG * 64 * 2);
  ushort_t* epool_bf = (ushort_t*)alloc((size_t)NG * 64 * 2);
  // zero-padded bf16 copies of raw inputs
  ushort_t* xpad  = (ushort_t*)alloc((size_t)NN * 96 * 2);   // K=92 -> 96
  ushort_t* eapad = (ushort_t*)alloc((size_t)NE * 64 * 2);   // K=50 -> 64
  ushort_t* upad  = (ushort_t*)alloc((size_t)NG * 32 * 2);   // K=3  -> 32

  // ----- forward pass -----
  k_gather_int<<<gblocks(NE), 256, 0, stream>>>(batch, row, ubr, NE);
  k_cvt_pad<<<gblocks((long long)NN * 96), 256, 0, stream>>>(x, 92, 96, xpad, NN);
  k_cvt_pad<<<gblocks((long long)NE * 64), 256, 0, stream>>>(edge_attr, 50, 64, eapad, NE);
  k_cvt_pad<<<gblocks((long long)NG * 32), 256, 0, stream>>>(u, 3, 32, upad, NG);

  // pre: xc = relu(x @ Wpre + bpre)   [NN,92] -> [NN,64]
  G1(stream, xpad, NN, 92, 96, pre.W, pre.b, xc, xc_bf, 1);

  const ushort_t* ecur = eapad; int eK = 50; int eStride = 64;
  const ushort_t* ucur = upad;  int uK = 3;  int uStride = 32;

  for (int i = 0; i < 3; ++i) {
    GCB& P = gc[i];
    // embeds (Linear->relu twice)
    G1(stream, ecur, NE, eK, eStride, P.e0.W, P.e0.b, eB, eB_bf, 1);
    G1(stream, eB_bf, NE, 64, 64, P.e1.W, P.e1.b, e_t, e_t_bf, 1);
    G1(stream, xc_bf, NN, 64, 64, P.x0.W, P.x0.b, nB, nB_bf, 1);
    G1(stream, nB_bf, NN, 64, 64, P.x1.W, P.x1.b, x_t, x_t_bf, 1);
    G1(stream, ucur, NG, uK, uStride, P.u0.W, P.u0.b, gB, gB_bf, 1);
    G1(stream, gB_bf, NG, 64, 64, P.u1.W, P.u1.b, u_t, u_t_bf, 1);

    // edge MLP on fused concat [x_t[row], x_t[col], e_t, u_t[batch[row]]]
    launch_gemm(stream, x_t_bf, row, x_t_bf, col, e_t_bf, nullptr, u_t_bf, ubr,
                4, NE, 256, 64, P.edge.lin[0].W, P.edge.lin[0].b, eA, nullptr, 1);
    run_bn(stream, eA, NE, P.edge.g[0], P.edge.be[0], accum, eA_bf);
    G1(stream, eA_bf, NE, 64, 64, P.edge.lin[1].W, P.edge.lin[1].b, eB, nullptr, 1);
    run_bn(stream, eB, NE, P.edge.g[1], P.edge.be[1], accum, eB_bf);
    G1(stream, eB_bf, NE, 64, 64, P.edge.lin[2].W, P.edge.lin[2].b, eA, nullptr, 1);
    run_bn(stream, eA, NE, P.edge.g[2], P.edge.be[2], accum, eA_bf);   // e_out = eA

    // v_e = smean(e_out, row, NN)  (result in vacc)
    run_smean(stream, eA, row, NE, vacc, cnt_n, NN, vacc_bf);

    // node MLP on fused concat [x_t, v_e, u_t[batch]]
    launch_gemm(stream, x_t_bf, nullptr, vacc_bf, nullptr, u_t_bf, batch,
                nullptr, nullptr, 3, NN, 192, 64,
                P.node.lin[0].W, P.node.lin[0].b, nA, nullptr, 1);
    run_bn(stream, nA, NN, P.node.g[0], P.node.be[0], accum, nA_bf);
    G1(stream, nA_bf, NN, 64, 64, P.node.lin[1].W, P.node.lin[1].b, nB, nullptr, 1);
    run_bn(stream, nB, NN, P.node.g[1], P.node.be[1], accum, nB_bf);
    G1(stream, nB_bf, NN, 64, 64, P.node.lin[2].W, P.node.lin[2].b, nA, nullptr, 1);
    run_bn(stream, nA, NN, P.node.g[2], P.node.be[2], accum, nA_bf);   // x_out = nA

    // u_e = smean(v_e, batch, NG); u_v = smean(x_out, batch, NG)
    run_smean(stream, vacc, batch, NN, gue, cnt_g, NG, gue_bf);
    run_smean(stream, nA,   batch, NN, guv, cnt_g, NG, guv_bf);

    // glob MLP on [u_e, u_v, u_t]
    launch_gemm(stream, gue_bf, nullptr, guv_bf, nullptr, u_t_bf, nullptr,
                nullptr, nullptr, 3, NG, 192, 64,
                P.glob.lin[0].W, P.glob.lin[0].b, gA, nullptr, 1);
    run_bn(stream, gA, NG, P.glob.g[0], P.glob.be[0], accum, gA_bf);
    G1(stream, gA_bf, NG, 64, 64, P.glob.lin[1].W, P.glob.lin[1].b, gB, nullptr, 1);
    run_bn(stream, gB, NG, P.glob.g[1], P.glob.be[1], accum, gB_bf);
    G1(stream, gB_bf, NG, 64, 64, P.glob.lin[2].W, P.glob.lin[2].b, gA, nullptr, 1);
    run_bn(stream, gA, NG, P.glob.g[2], P.glob.be[2], accum, gA_bf);   // u_out = gA

    // residuals (f32 + bf16 shadow)
    k_add<<<gblocks((long long)NE * 64), 256, 0, stream>>>(ec, eA, i ? ec : e_t, ec_bf, (long long)NE * 64);
    k_add<<<gblocks((long long)NN * 64), 256, 0, stream>>>(xc, nA, i ? xc : x_t, xc_bf, (long long)NN * 64);
    k_add<<<gblocks((long long)NG * 64), 256, 0, stream>>>(ucb, gA, i ? ucb : u_t, ucb_bf, (long long)NG * 64);

    ecur = ec_bf; eK = 64; eStride = 64;
    ucur = ucb_bf; uK = 64; uStride = 64;
  }

  // pools
  run_smean(stream, xc, batch, NN, xpool, cnt_g, NG, xpool_bf);
  run_smean(stream, ec, row, NE, vacc, cnt_n, NN, vacc_bf);
  run_smean(stream, vacc, batch, NN, epool, cnt_g, NG, epool_bf);

  // post: relu([x_pool, e_pool, uc] @ Wpost + bpost), then final linear
  launch_gemm(stream, xpool_bf, nullptr, epool_bf, nullptr, ucb_bf, nullptr,
              nullptr, nullptr, 3, NG, 192, 64, post.W, post.b, pA, nullptr, 1);
  k_final<<<1, 256, 0, stream>>>(pA, outl.W, outl.b, (float*)d_out);
}
